// AgentAttention_2757369004151
// MI455X (gfx1250) — matrix operand order repaired and checked by reading, not compiled, not run on hardware
//
#include <hip/hip_runtime.h>
#include <cstddef>
#include <cstdint>

// ---------------------------------------------------------------------------
// Agent Attention on MI455X (gfx1250), bf16 WMMA pipeline.
//   B=4, N=4096, NA=64, DIM=1024, H=16, D=64, SCALE=0.125
// Async global->LDS copies (ASYNCcnt) + double-buffered GEMM K-loop.
// ---------------------------------------------------------------------------

typedef __attribute__((ext_vector_type(16))) __bf16 v16bf;
typedef __attribute__((ext_vector_type(8)))  __bf16 v8bf;
typedef __attribute__((ext_vector_type(4)))  __bf16 v4bf;
typedef __attribute__((ext_vector_type(8)))  float  v8f;
typedef __attribute__((ext_vector_type(4)))  int    v4i;

#define DIM_C   1024
#define NTOK    4096
#define NB      4
#define NA_C    64
#define HEADS   16
#define HDIM    64
#define SCALE_C 0.125f

// ---- async global -> LDS copy (CDNA5 path, guarded) -----------------------

#if __has_builtin(__builtin_amdgcn_global_load_async_to_lds_b128)
#define USE_ASYNC_COPY 1
#else
#define USE_ASYNC_COPY 0
#endif

typedef __attribute__((address_space(1))) v4i as1_v4i;
typedef __attribute__((address_space(3))) v4i as3_v4i;

__device__ __forceinline__ void cp_async16(__bf16* lds_dst, const __bf16* g_src) {
#if USE_ASYNC_COPY
    __builtin_amdgcn_global_load_async_to_lds_b128(
        (as1_v4i*)g_src, (as3_v4i*)lds_dst, 0, 0);
#else
    *(v8bf*)lds_dst = *(const v8bf*)g_src;
#endif
}

__device__ __forceinline__ void cp_async_wait() {
#if USE_ASYNC_COPY
#if __has_builtin(__builtin_amdgcn_s_wait_asynccnt)
    __builtin_amdgcn_s_wait_asynccnt(0);
#else
    asm volatile("s_wait_asynccnt 0x0" ::: "memory");
#endif
#endif
}

// ---- WMMA helpers ---------------------------------------------------------

__device__ __forceinline__ v8f wmma_bf16(v16bf a, v16bf b, v8f c) {
    // (neg_a, A, neg_b, B, c_mod, C, reuse_a, reuse_b)
    return __builtin_amdgcn_wmma_f32_16x16x32_bf16(false, a, false, b,
                                                   (short)0, c, false, false);
}

// A-fragment: lane holds row M; v16bf elems 0..15 = K[off..off+15];
// caller folds (lane&16 ? 16 : 0) into p.
__device__ __forceinline__ v16bf frag_split(const __bf16* p) {
    v8bf lo = *(const v8bf*)(p);
    v8bf hi = *(const v8bf*)(p + 8);
    return __builtin_shufflevector(lo, hi, 0,1,2,3,4,5,6,7,8,9,10,11,12,13,14,15);
}

// B-fragment: lane holds column N; 16 contiguous K values
// (caller folds (lane&16 ? 16 : 0) into p).
__device__ __forceinline__ v16bf frag_contig(const __bf16* p) {
    v8bf lo = *(const v8bf*)(p);
    v8bf hi = *(const v8bf*)(p + 8);
    return __builtin_shufflevector(lo, hi, 0,1,2,3,4,5,6,7,8,9,10,11,12,13,14,15);
}

// ---- f32 -> bf16 cast -----------------------------------------------------

__global__ __launch_bounds__(256) void cast_f32_bf16(const float* __restrict__ in,
                                                     __bf16* __restrict__ out,
                                                     int n4) {
    int i = blockIdx.x * 256 + threadIdx.x;
    if (i < n4) {
        float4 f = ((const float4*)in)[i];
        v4bf o;
        o.x = (__bf16)f.x; o.y = (__bf16)f.y;
        o.z = (__bf16)f.z; o.w = (__bf16)f.w;
        ((v4bf*)out)[i] = o;
    }
}

// ---- Tiled WMMA GEMM: Y[M,Nout] = X[M,K] @ W[Nout,K]^T --------------------
// Block tile 128x128, 8 waves (4x2), wave tile 32x64, K-step 32.
// Double-buffered LDS with async copies: one barrier per K-step.

template<bool OUT_BF16>
__global__ __launch_bounds__(256) void gemm_wmma(const __bf16* __restrict__ X,
                                                 const __bf16* __restrict__ W,
                                                 __bf16* __restrict__ Ybf,
                                                 float*  __restrict__ Yf,
                                                 const float* __restrict__ bias,
                                                 int M, int K, int Nout) {
    const int LDT = 40;                       // padded LDS stride (bf16)
    __shared__ __bf16 As[2][128 * LDT];
    __shared__ __bf16 Bs[2][128 * LDT];

    const int tid  = threadIdx.x;
    const int lane = tid & 31, wid = tid >> 5;
    const int waveM = wid >> 1, waveN = wid & 1;
    const int m0 = blockIdx.y * 128, n0 = blockIdx.x * 128;
    const int nlane = lane & 15;
    const int mhi   = (lane & 16) ? 8 : 0;
    const int krow  = (lane & 16) ? 16 : 0;   // A contiguous offset
    const int kb16  = (lane & 16) ? 16 : 0;   // B contiguous offset

    // two 16B segments per thread per matrix (512 segs of 16B each)
    const int r0 = tid >> 2,         c0 = (tid & 3) * 8;
    const int r1 = (tid + 256) >> 2, c1 = ((tid + 256) & 3) * 8;

    auto load_tile = [&](int buf, int k0) {
        cp_async16(&As[buf][r0 * LDT + c0], X + (size_t)(m0 + r0) * K + k0 + c0);
        cp_async16(&As[buf][r1 * LDT + c1], X + (size_t)(m0 + r1) * K + k0 + c1);
        cp_async16(&Bs[buf][r0 * LDT + c0], W + (size_t)(n0 + r0) * K + k0 + c0);
        cp_async16(&Bs[buf][r1 * LDT + c1], W + (size_t)(n0 + r1) * K + k0 + c1);
    };

    v8f acc[2][4] = {};

    load_tile(0, 0);                          // prologue
    cp_async_wait();
    __syncthreads();

    int cur = 0;
    for (int k0 = 0; k0 < K; k0 += 32) {
        const int nxt = cur ^ 1;
        if (k0 + 32 < K) load_tile(nxt, k0 + 32);   // overlap with WMMA below

        v16bf af[2], bfr[4];
#pragma unroll
        for (int mt = 0; mt < 2; ++mt)
            af[mt] = frag_split(&As[cur][(waveM * 32 + mt * 16 + nlane) * LDT + krow]);
#pragma unroll
        for (int nt = 0; nt < 4; ++nt)
            bfr[nt] = frag_contig(&Bs[cur][(waveN * 64 + nt * 16 + nlane) * LDT + kb16]);
#pragma unroll
        for (int mt = 0; mt < 2; ++mt)
#pragma unroll
            for (int nt = 0; nt < 4; ++nt)
                acc[mt][nt] = wmma_bf16(af[mt], bfr[nt], acc[mt][nt]);

        cp_async_wait();                      // next tile landed in LDS
        __syncthreads();                      // everyone done reading 'cur'
        cur = nxt;
    }

#pragma unroll
    for (int mt = 0; mt < 2; ++mt) {
#pragma unroll
        for (int nt = 0; nt < 4; ++nt) {
            const int col = n0 + waveN * 64 + nt * 16 + nlane;
            float bv = 0.f;
            if (!OUT_BF16) bv = bias[col];
#pragma unroll
            for (int r = 0; r < 8; ++r) {
                const int row = m0 + waveM * 32 + mt * 16 + r + mhi;
                const float v = acc[mt][nt][r];
                if (OUT_BF16) Ybf[(size_t)row * Nout + col] = (__bf16)v;
                else          Yf [(size_t)row * Nout + col] = v + bv;
            }
        }
    }
}

// ---- Stage 1: flash agent->keys attention, one WG per (b,h) ---------------
// agent_v stored TRANSPOSED: AVt[b,h][dim][agent]  (B-operand-ready).

__global__ __launch_bounds__(256) void agent_stage1(const __bf16* __restrict__ Abf,
                                                    const __bf16* __restrict__ Kbf,
                                                    const __bf16* __restrict__ Vbf,
                                                    __bf16* __restrict__ AVt) {
    const int b = blockIdx.x >> 4, h = blockIdx.x & 15;
    const int LP = 72;                        // padded bf16 stride
    __shared__ __bf16 a_s [64 * LP];
    __shared__ __bf16 k_s [64 * LP];
    __shared__ __bf16 vt_s[64 * LP];          // transposed V chunk: [dim][key]
    __shared__ __bf16 p_s [64 * LP];
    __shared__ float  S_s [64 * 64];
    __shared__ float  m_s[64], l_s[64], al_s[64];

    const int tid = threadIdx.x, lane = tid & 31, wid = tid >> 5;
    const int nlane = lane & 15;
    const int mhi  = (lane & 16) ? 8 : 0;
    const int krow = (lane & 16) ? 16 : 0;
    const int kb16 = (lane & 16) ? 16 : 0;
    const int mi    = wid >> 1;               // agent-row tile 0..3
    const int npair = (wid & 1) * 2;          // col tiles {npair, npair+1}

    const __bf16* Ab = Abf + (size_t)b * NA_C * DIM_C + h * HDIM;
    const __bf16* Kb = Kbf + (size_t)b * NTOK * DIM_C + h * HDIM;
    const __bf16* Vb = Vbf + (size_t)b * NTOK * DIM_C + h * HDIM;

#pragma unroll
    for (int i = 0; i < 2; ++i) {             // agent tile: 64x64 bf16 (async)
        int seg = tid + i * 256;
        int row = seg >> 3, c = (seg & 7) * 8;
        cp_async16(a_s + row * LP + c, Ab + (size_t)row * DIM_C + c);
    }
    if (tid < 64) { m_s[tid] = -1e30f; l_s[tid] = 0.f; }
    cp_async_wait();
    __syncthreads();

    v16bf afrag[2];
#pragma unroll
    for (int ks = 0; ks < 2; ++ks)
        afrag[ks] = frag_split(a_s + (mi * 16 + nlane) * LP + ks * 32 + krow);

    v8f o[2] = {};

    for (int ch = 0; ch < NTOK / 64; ++ch) {
#pragma unroll
        for (int i = 0; i < 2; ++i) {         // K async row-major, V transposed
            int seg = tid + i * 256;
            int row = seg >> 3, c = (seg & 7) * 8;
            const size_t gr = (size_t)(ch * 64 + row) * DIM_C;
            cp_async16(k_s + row * LP + c, Kb + gr + c);
            v8bf vv = *(const v8bf*)(Vb + gr + c);
#pragma unroll
            for (int e = 0; e < 8; ++e) vt_s[(c + e) * LP + row] = vv[e];
        }
        cp_async_wait();
        __syncthreads();

#pragma unroll
        for (int j = 0; j < 2; ++j) {         // S = a @ K^T  (2 tiles/wave)
            const int ni = npair + j;
            v8f s = {};
#pragma unroll
            for (int ks = 0; ks < 2; ++ks) {
                v16bf kf = frag_contig(k_s + (ni * 16 + nlane) * LP + ks * 32 + kb16);
                s = wmma_bf16(afrag[ks], kf, s);
            }
#pragma unroll
            for (int r = 0; r < 8; ++r)
                S_s[(mi * 16 + r + mhi) * 64 + ni * 16 + nlane] = s[r] * SCALE_C;
        }
        __syncthreads();

        if (tid < 64) {                       // online softmax update, row=tid
            const int row = tid;
            float mold = m_s[row], cmax = mold;
            for (int c = 0; c < 64; ++c) cmax = fmaxf(cmax, S_s[row * 64 + c]);
            const float alpha = __expf(mold - cmax);
            float sum = 0.f;
            for (int c = 0; c < 64; ++c) {
                const float p = __expf(S_s[row * 64 + c] - cmax);
                sum += p;
                p_s[row * LP + c] = (__bf16)p;
            }
            l_s[row] = l_s[row] * alpha + sum;
            m_s[row] = cmax;
            al_s[row] = alpha;
        }
        __syncthreads();

        v16bf pf[2];
#pragma unroll
        for (int ks = 0; ks < 2; ++ks)
            pf[ks] = frag_split(p_s + (mi * 16 + nlane) * LP + ks * 32 + krow);
#pragma unroll
        for (int j = 0; j < 2; ++j) {         // o = o*alpha + P @ V
            const int ni = npair + j;
#pragma unroll
            for (int r = 0; r < 8; ++r)
                o[j][r] *= al_s[mi * 16 + r + mhi];
#pragma unroll
            for (int ks = 0; ks < 2; ++ks) {
                v16bf vf = frag_contig(vt_s + (ni * 16 + nlane) * LP + ks * 32 + kb16);
                o[j] = wmma_bf16(pf[ks], vf, o[j]);
            }
        }
        __syncthreads();
    }

    __bf16* dst = AVt + (size_t)(b * HEADS + h) * NA_C * HDIM;  // [dim][agent]
#pragma unroll
    for (int j = 0; j < 2; ++j) {
        const int ni = npair + j;
#pragma unroll
        for (int r = 0; r < 8; ++r) {
            const int ag = mi * 16 + r + mhi;
            const int d  = ni * 16 + nlane;
            dst[(size_t)d * NA_C + ag] = (__bf16)(o[j][r] / l_s[ag]);
        }
    }
}

// ---- Stage 2: queries attend over 64 agents, 128 queries per WG -----------

__global__ __launch_bounds__(256) void agent_stage2(const __bf16* __restrict__ Qbf,
                                                    const __bf16* __restrict__ Abf,
                                                    const __bf16* __restrict__ AVt,
                                                    __bf16* __restrict__ Obf) {
    int idx = blockIdx.x;
    const int qt = idx & 31; idx >>= 5;
    const int h  = idx & 15; const int b = idx >> 4;

    __shared__ __bf16 q_s[128 * 64];          // reused for P after softmax
    __shared__ __bf16 a_s[64 * 64];
    __shared__ float  S_s[128 * 64];          // start reused for avT bf16
    __bf16* avt_s = (__bf16*)S_s;

    const int tid = threadIdx.x, lane = tid & 31, wid = tid >> 5;
    const int nlane = lane & 15;
    const int mhi  = (lane & 16) ? 8 : 0;
    const int krow = (lane & 16) ? 16 : 0;
    const int kb16 = (lane & 16) ? 16 : 0;
    const int mi = wid;                       // 16 query rows per wave

    const __bf16* Qb = Qbf + ((size_t)(b * NTOK + qt * 128)) * DIM_C + h * HDIM;
    const __bf16* Ab = Abf + (size_t)b * NA_C * DIM_C + h * HDIM;

#pragma unroll
    for (int i = 0; i < 4; ++i) {             // 128x64 q tile (async)
        int seg = tid + i * 256;
        int row = seg >> 3, c = (seg & 7) * 8;
        cp_async16(q_s + row * 64 + c, Qb + (size_t)row * DIM_C + c);
    }
#pragma unroll
    for (int i = 0; i < 2; ++i) {             // 64x64 agent tile (async)
        int seg = tid + i * 256;
        int row = seg >> 3, c = (seg & 7) * 8;
        cp_async16(a_s + row * 64 + c, Ab + (size_t)row * DIM_C + c);
    }
    cp_async_wait();
    __syncthreads();

    {                                         // S = q @ a^T
        v16bf qf[2];
#pragma unroll
        for (int ks = 0; ks < 2; ++ks)
            qf[ks] = frag_split(q_s + (mi * 16 + nlane) * 64 + ks * 32 + krow);
#pragma unroll
        for (int ni = 0; ni < 4; ++ni) {
            v8f s = {};
#pragma unroll
            for (int ks = 0; ks < 2; ++ks) {
                v16bf af = frag_contig(a_s + (ni * 16 + nlane) * 64 + ks * 32 + kb16);
                s = wmma_bf16(qf[ks], af, s);
            }
#pragma unroll
            for (int r = 0; r < 8; ++r)
                S_s[(mi * 16 + r + mhi) * 64 + ni * 16 + nlane] = s[r] * SCALE_C;
        }
    }
    __syncthreads();

    if (tid < 128) {                          // full softmax over 64 agents
        const int row = tid;
        float cmax = -1e30f;
        for (int c = 0; c < 64; ++c) cmax = fmaxf(cmax, S_s[row * 64 + c]);
        float sum = 0.f;
        float e[64];
        for (int c = 0; c < 64; ++c) { e[c] = __expf(S_s[row * 64 + c] - cmax); sum += e[c]; }
        const float inv = 1.f / sum;
        for (int c = 0; c < 64; ++c) q_s[row * 64 + c] = (__bf16)(e[c] * inv);
    }
    __syncthreads();

#pragma unroll
    for (int i = 0; i < 2; ++i) {             // transposed agent_v (S_s dead)
        int seg = tid + i * 256;
        int row = seg >> 3, c = (seg & 7) * 8;
        const __bf16* src = AVt + (size_t)(b * HEADS + h) * NA_C * HDIM;
        cp_async16(avt_s + row * 64 + c, src + (size_t)row * NA_C + c);
    }
    cp_async_wait();
    __syncthreads();

    {                                         // out = P @ agent_v
        v16bf pf[2];
#pragma unroll
        for (int ks = 0; ks < 2; ++ks)
            pf[ks] = frag_split(q_s + (mi * 16 + nlane) * 64 + ks * 32 + krow);
#pragma unroll
        for (int ni = 0; ni < 4; ++ni) {
            v8f ov = {};
#pragma unroll
            for (int ks = 0; ks < 2; ++ks) {
                v16bf vf = frag_contig(avt_s + (ni * 16 + nlane) * 64 + ks * 32 + kb16);
                ov = wmma_bf16(pf[ks], vf, ov);
            }
#pragma unroll
            for (int r = 0; r < 8; ++r) {
                const int qrow = b * NTOK + qt * 128 + mi * 16 + r + mhi;
                const int col  = h * HDIM + ni * 16 + nlane;
                Obf[(size_t)qrow * DIM_C + col] = (__bf16)ov[r];
            }
        }
    }
}

// ---------------------------------------------------------------------------

extern "C" void kernel_launch(void* const* d_in, const int* in_sizes, int n_in,
                              void* d_out, int out_size, void* d_ws, size_t ws_size,
                              hipStream_t stream) {
    const float* x     = (const float*)d_in[0];
    const float* agent = (const float*)d_in[1];
    const float* Wq    = (const float*)d_in[2];
    const float* Wk    = (const float*)d_in[3];
    const float* Wv    = (const float*)d_in[4];
    const float* Wp    = (const float*)d_in[5];
    const float* bp    = (const float*)d_in[6];
    float* out = (float*)d_out;
    char* ws = (char*)d_ws;

    const size_t MB = 1u << 20;
    __bf16* wq_bf = (__bf16*)(ws + 0 * MB);
    __bf16* wk_bf = (__bf16*)(ws + 2 * MB);
    __bf16* wv_bf = (__bf16*)(ws + 4 * MB);
    __bf16* wp_bf = (__bf16*)(ws + 6 * MB);
    __bf16* ag_bf = (__bf16*)(ws + 8 * MB);
    __bf16* avt   = (__bf16*)(ws + 9 * MB);
    __bf16* x_bf  = (__bf16*)(ws + 16 * MB);   // reused as attn_out after QKV
    __bf16* q_bf  = (__bf16*)(ws + 48 * MB);
    __bf16* k_bf  = (__bf16*)(ws + 80 * MB);
    __bf16* v_bf  = (__bf16*)(ws + 112 * MB);
    __bf16* attn_bf = x_bf;

    const int M = NB * NTOK;                   // 16384
    const int wn4 = DIM_C * DIM_C / 4;         // 262144
    const int xn4 = M * DIM_C / 4;             // 4194304
    const int an4 = NB * NA_C * DIM_C / 4;     // 65536

    // f32 -> bf16
    cast_f32_bf16<<<(wn4 + 255) / 256, 256, 0, stream>>>(Wq, wq_bf, wn4);
    cast_f32_bf16<<<(wn4 + 255) / 256, 256, 0, stream>>>(Wk, wk_bf, wn4);
    cast_f32_bf16<<<(wn4 + 255) / 256, 256, 0, stream>>>(Wv, wv_bf, wn4);
    cast_f32_bf16<<<(wn4 + 255) / 256, 256, 0, stream>>>(Wp, wp_bf, wn4);
    cast_f32_bf16<<<(xn4 + 255) / 256, 256, 0, stream>>>(x, x_bf, xn4);
    cast_f32_bf16<<<(an4 + 255) / 256, 256, 0, stream>>>(agent, ag_bf, an4);

    // Q/K/V projections (bf16 out)
    dim3 ggrid(DIM_C / 128, M / 128);
    gemm_wmma<true><<<ggrid, 256, 0, stream>>>(x_bf, wq_bf, q_bf, nullptr, nullptr, M, DIM_C, DIM_C);
    gemm_wmma<true><<<ggrid, 256, 0, stream>>>(x_bf, wk_bf, k_bf, nullptr, nullptr, M, DIM_C, DIM_C);
    gemm_wmma<true><<<ggrid, 256, 0, stream>>>(x_bf, wv_bf, v_bf, nullptr, nullptr, M, DIM_C, DIM_C);

    // Stage 1: agents attend over keys (flash, one WG per (b,h))
    agent_stage1<<<NB * HEADS, 256, 0, stream>>>(ag_bf, k_bf, v_bf, avt);

    // Stage 2: queries attend over agents (x_bf region now reused)
    agent_stage2<<<NB * HEADS * (NTOK / 128), 256, 0, stream>>>(q_bf, ag_bf, avt, attn_bf);

    // Output projection with bias (f32 out)
    gemm_wmma<false><<<ggrid, 256, 0, stream>>>(attn_bf, wp_bf, nullptr, out, bp, M, DIM_C, DIM_C);
}
